// DeepseekV2MoETileGym_52879637348905
// MI455X (gfx1250) — compile-verified, matrix operand order checked by reading, not run on hardware
//
#include <hip/hip_runtime.h>

// ---------------- problem constants ----------------
#define T_TOK 2048
#define HDIM  2048
#define IDIM  1408
#define NEXP  16
#define TOPK  6
#define ISDIM 2816   // shared-expert intermediate

// ---------------- types ----------------
typedef __attribute__((ext_vector_type(16))) __bf16         bf16x16;
typedef __attribute__((ext_vector_type(2)))  __bf16         bf16x2;
typedef __attribute__((ext_vector_type(8)))  float          f32x8;
typedef __attribute__((ext_vector_type(8)))  unsigned short u16x8;
typedef __attribute__((ext_vector_type(4)))  int            v4i;

struct BFragBits { u16x8 lo, hi; };

// gfx1250 async copy to LDS (ASYNCcnt-tracked), guarded so compile never breaks.
#if defined(__AMDGCN__) && __has_builtin(__builtin_amdgcn_global_load_async_to_lds_b128) && \
    __has_builtin(__builtin_amdgcn_s_wait_asynccnt)
#define USE_ASYNC_LDS 1
#else
#define USE_ASYNC_LDS 0
#endif

#if USE_ASYNC_LDS
// signature (probed via hipcc diagnostic): (v4i* gsrc, v4i* lds, imm offset, imm cpol)
#define STAGE_A_ASYNC(gsrc, ldst)                                              \
    __builtin_amdgcn_global_load_async_to_lds_b128(                            \
        reinterpret_cast<v4i*>(const_cast<unsigned short*>(gsrc)),             \
        reinterpret_cast<v4i*>(ldst), 0, 0)
#define WAIT_ASYNC() __builtin_amdgcn_s_wait_asynccnt(0)
#endif

// fp32 -> bf16 via native hardware convert (RNE), not manual bit-twiddling:
// lets clang select v_cvt_pk_bf16_f32 instead of mov_b16/and/add3 chains.
__device__ __forceinline__ unsigned short f2bf(float f) {
    return __builtin_bit_cast(unsigned short, (__bf16)f);
}
__device__ __forceinline__ unsigned int pack2(float a, float b) {
    bf16x2 v; v.x = (__bf16)a; v.y = (__bf16)b;
    return __builtin_bit_cast(unsigned int, v);
}

// Fragment gather from an LDS tile (rows contiguous in K, stride LDS_PAD).
// CDNA5 16-bit operand layout: lanes 0-15 hold K[0..8)+K[16..24),
// lanes 16-31 hold K[8..16)+K[24..32).
#define LDS_PAD 40   // 80B rows: 16B-aligned b128 loads, conflict-free banks
__device__ __forceinline__ bf16x16 lds_frag(const unsigned short* base, int row, int k0) {
    const unsigned short* p = base + row * LDS_PAD + k0;
    BFragBits f;
    f.lo = *reinterpret_cast<const u16x8*>(p);
    f.hi = *reinterpret_cast<const u16x8*>(p + 16);
    return __builtin_bit_cast(bf16x16, f);
}

// ================= gate: softmax + top-6 combine weights =================
__global__ __launch_bounds__(256) void moe_gate(const float* __restrict__ x,
                                                const float* __restrict__ gate_w,
                                                float* __restrict__ combine) {
    const int lane = threadIdx.x & 31;
    const int wv   = threadIdx.x >> 5;
    const int t    = blockIdx.x * 8 + wv;

    float s[NEXP];
#pragma unroll
    for (int e = 0; e < NEXP; ++e) s[e] = 0.f;

    const float* xr = x + (size_t)t * HDIM;
    for (int k = lane; k < HDIM; k += 32) {
        float xv = xr[k];
#pragma unroll
        for (int e = 0; e < NEXP; ++e) s[e] += xv * gate_w[e * HDIM + k];
    }
#pragma unroll
    for (int off = 16; off > 0; off >>= 1) {
#pragma unroll
        for (int e = 0; e < NEXP; ++e) s[e] += __shfl_xor(s[e], off);
    }
    float m = s[0];
#pragma unroll
    for (int e = 1; e < NEXP; ++e) m = fmaxf(m, s[e]);
    float d = 0.f;
#pragma unroll
    for (int e = 0; e < NEXP; ++e) { s[e] = __expf(s[e] - m); d += s[e]; }
    float inv = 1.f / d;
#pragma unroll
    for (int e = 0; e < NEXP; ++e) s[e] *= inv;

    unsigned sel = 0; float wsum = 0.f;
#pragma unroll
    for (int it = 0; it < TOPK; ++it) {
        float bv = -1e30f; int bi = 0;
#pragma unroll
        for (int e = 0; e < NEXP; ++e) {
            float v = ((sel >> e) & 1u) ? -1e30f : s[e];
            if (v > bv) { bv = v; bi = e; }
        }
        sel |= 1u << bi; wsum += bv;
    }
    float winv = 1.f / (wsum + 1e-20f);
    if (lane == 0) {
#pragma unroll
        for (int e = 0; e < NEXP; ++e)
            combine[(size_t)t * NEXP + e] = ((sel >> e) & 1u) ? s[e] * winv : 0.f;
    }
}

// ================= fp32 -> bf16 activation convert =================
__global__ __launch_bounds__(256) void cvt_bf16(const float4* __restrict__ x,
                                                uint2* __restrict__ xb, int n4) {
    int i = blockIdx.x * blockDim.x + threadIdx.x;
    if (i < n4) {
        float4 v = x[i];
        uint2 o; o.x = pack2(v.x, v.y); o.y = pack2(v.z, v.w);
        xb[i] = o;
    }
}

// ================= zero output =================
__global__ __launch_bounds__(256) void zero_f32(float4* __restrict__ p, int n4) {
    int i = blockIdx.x * blockDim.x + threadIdx.x;
    if (i < n4) p[i] = make_float4(0.f, 0.f, 0.f, 0.f);
}

// ================= GEMM1: gu = x @ w13^T, fused SwiGLU -> bf16 act =========
// 64 tokens x 128 cols; wave w owns cols [w*16,w*16+16) for BOTH g and u.
// Double-buffered LDS pipeline, one barrier per 32-K step; A tile staged by
// async-copy-to-LDS (ASYNCcnt).
__global__ __launch_bounds__(256) void moe_gemm1(const unsigned short* __restrict__ xb, // [T,Kdim] bf16
                                                 const float* __restrict__ w13,         // g=f, u=uOff+f
                                                 unsigned short* __restrict__ act,      // [T,Fdim] bf16
                                                 int Kdim, int Fdim, int uOff) {
    __shared__ __align__(16) unsigned short As[2][64 * LDS_PAD];
    __shared__ __align__(16) unsigned short Bs[2][256 * LDS_PAD];

    const int tid  = threadIdx.x;
    const int lane = tid & 31;
    const int wv   = tid >> 5;
    const int m0   = blockIdx.y * 64;
    const int f0   = blockIdx.x * 128;

    f32x8 accg[4], accu[4];
    const f32x8 z = {0.f, 0.f, 0.f, 0.f, 0.f, 0.f, 0.f, 0.f};
#pragma unroll
    for (int i = 0; i < 4; ++i) { accg[i] = z; accu[i] = z; }

    const int arow = tid >> 2, aseg = (tid & 3) * 8;     // A: 16B per thread
    const unsigned short* xrow = xb + (size_t)(m0 + arow) * Kdim + aseg;
    const int brow = tid;                                // B: 32 fp32 per thread
    const int gcol = (brow < 128) ? (f0 + brow) : (uOff + f0 + brow - 128);
    const float* wrow = w13 + (size_t)gcol * Kdim;

    const int rA = lane & 15;
    const int k0 = (lane < 16) ? 0 : 8;
    const int nsteps = Kdim >> 5;

    // ---- prologue: stage tile 0 into buffer 0 ----
#if USE_ASYNC_LDS
    STAGE_A_ASYNC(xrow, &As[0][arow * LDS_PAD + aseg]);
#else
    *reinterpret_cast<uint4*>(&As[0][arow * LDS_PAD + aseg]) =
        *reinterpret_cast<const uint4*>(xrow);
#endif
#pragma unroll
    for (int s = 0; s < 4; ++s) {
        float4 fa = *reinterpret_cast<const float4*>(wrow + s * 8);
        float4 fb = *reinterpret_cast<const float4*>(wrow + s * 8 + 4);
        *reinterpret_cast<uint4*>(&Bs[0][brow * LDS_PAD + s * 8]) =
            make_uint4(pack2(fa.x, fa.y), pack2(fa.z, fa.w),
                       pack2(fb.x, fb.y), pack2(fb.z, fb.w));
    }
#if USE_ASYNC_LDS
    WAIT_ASYNC();
#endif
    __syncthreads();

    int cur = 0;
    for (int ks = 0; ks < nsteps; ++ks) {
        const int nxt = cur ^ 1;
        const bool more = (ks + 1) < nsteps;
        const int kn = (ks + 1) << 5;

        // ---- issue next-tile loads (overlapped with WMMA below) ----
        float4 pfa[4], pfb[4];
#if !USE_ASYNC_LDS
        uint4 areg;
#endif
        if (more) {
#if USE_ASYNC_LDS
            STAGE_A_ASYNC(xrow + kn, &As[nxt][arow * LDS_PAD + aseg]);
#else
            areg = *reinterpret_cast<const uint4*>(xrow + kn);
#endif
#pragma unroll
            for (int s = 0; s < 4; ++s) {
                pfa[s] = *reinterpret_cast<const float4*>(wrow + kn + s * 8);
                pfb[s] = *reinterpret_cast<const float4*>(wrow + kn + s * 8 + 4);
            }
        }

        // ---- compute on current buffer ----
        const unsigned short* Ac = As[cur];
        const unsigned short* Bc = Bs[cur];
        bf16x16 bg = lds_frag(Bc, wv * 16 + rA, k0);
        bf16x16 bu = lds_frag(Bc, 128 + wv * 16 + rA, k0);
#pragma unroll
        for (int mi = 0; mi < 4; ++mi) {
            bf16x16 a = lds_frag(Ac, mi * 16 + rA, k0);
            accg[mi] = __builtin_amdgcn_wmma_f32_16x16x32_bf16(false, a, false, bg,
                                                               (short)0, accg[mi], false, false);
            accu[mi] = __builtin_amdgcn_wmma_f32_16x16x32_bf16(false, a, false, bu,
                                                               (short)0, accu[mi], false, false);
        }

        // ---- commit next tile to its buffer, single barrier ----
        if (more) {
#if !USE_ASYNC_LDS
            *reinterpret_cast<uint4*>(&As[nxt][arow * LDS_PAD + aseg]) = areg;
#endif
#pragma unroll
            for (int s = 0; s < 4; ++s) {
                *reinterpret_cast<uint4*>(&Bs[nxt][brow * LDS_PAD + s * 8]) =
                    make_uint4(pack2(pfa[s].x, pfa[s].y), pack2(pfa[s].z, pfa[s].w),
                               pack2(pfb[s].x, pfb[s].y), pack2(pfb[s].z, pfb[s].w));
            }
#if USE_ASYNC_LDS
            WAIT_ASYNC();
#endif
            __syncthreads();
        }
        cur = nxt;
    }

    // ---- SwiGLU epilogue (g,u in-register) ----
    const int fcol  = f0 + wv * 16 + (lane & 15);
    const int mbase = m0 + ((lane < 16) ? 0 : 8);
#pragma unroll
    for (int mi = 0; mi < 4; ++mi) {
#pragma unroll
        for (int j = 0; j < 8; ++j) {
            int t   = mbase + mi * 16 + j;
            float g = accg[mi][j];
            float u = accu[mi][j];
            float a = (g / (1.f + __expf(-g))) * u;
            act[(size_t)t * Fdim + fcol] = f2bf(a);
        }
    }
}

// ================= GEMM2: out += combine[:,e] * (act @ w2^T) =================
__global__ __launch_bounds__(256) void moe_gemm2(const unsigned short* __restrict__ actb, // [T,Kdim] bf16
                                                 const float* __restrict__ w2,            // [H,Kdim] fp32
                                                 const float* __restrict__ combine,       // [T,E] or null
                                                 int expert,
                                                 float* __restrict__ out, int Kdim) {
    __shared__ __align__(16) unsigned short As[2][64 * LDS_PAD];
    __shared__ __align__(16) unsigned short Bs[2][128 * LDS_PAD];

    const int tid  = threadIdx.x;
    const int lane = tid & 31;
    const int wv   = tid >> 5;
    const int m0   = blockIdx.y * 64;
    const int n0   = blockIdx.x * 128;

    f32x8 acc[4];
    const f32x8 z = {0.f, 0.f, 0.f, 0.f, 0.f, 0.f, 0.f, 0.f};
#pragma unroll
    for (int i = 0; i < 4; ++i) acc[i] = z;

    const int arow = tid >> 2, aseg = (tid & 3) * 8;
    const unsigned short* xrow = actb + (size_t)(m0 + arow) * Kdim + aseg;
    const int brow = tid >> 1, bseg = (tid & 1) * 16;    // B: 16 fp32 per thread
    const float* wrow = w2 + (size_t)(n0 + brow) * Kdim + bseg;

    const int rA = lane & 15;
    const int k0 = (lane < 16) ? 0 : 8;
    const int nsteps = Kdim >> 5;

    // ---- prologue ----
#if USE_ASYNC_LDS
    STAGE_A_ASYNC(xrow, &As[0][arow * LDS_PAD + aseg]);
#else
    *reinterpret_cast<uint4*>(&As[0][arow * LDS_PAD + aseg]) =
        *reinterpret_cast<const uint4*>(xrow);
#endif
#pragma unroll
    for (int s = 0; s < 2; ++s) {
        float4 fa = *reinterpret_cast<const float4*>(wrow + s * 8);
        float4 fb = *reinterpret_cast<const float4*>(wrow + s * 8 + 4);
        *reinterpret_cast<uint4*>(&Bs[0][brow * LDS_PAD + bseg + s * 8]) =
            make_uint4(pack2(fa.x, fa.y), pack2(fa.z, fa.w),
                       pack2(fb.x, fb.y), pack2(fb.z, fb.w));
    }
#if USE_ASYNC_LDS
    WAIT_ASYNC();
#endif
    __syncthreads();

    int cur = 0;
    for (int ks = 0; ks < nsteps; ++ks) {
        const int nxt = cur ^ 1;
        const bool more = (ks + 1) < nsteps;
        const int kn = (ks + 1) << 5;

        float4 pfa[2], pfb[2];
#if !USE_ASYNC_LDS
        uint4 areg;
#endif
        if (more) {
#if USE_ASYNC_LDS
            STAGE_A_ASYNC(xrow + kn, &As[nxt][arow * LDS_PAD + aseg]);
#else
            areg = *reinterpret_cast<const uint4*>(xrow + kn);
#endif
#pragma unroll
            for (int s = 0; s < 2; ++s) {
                pfa[s] = *reinterpret_cast<const float4*>(wrow + kn + s * 8);
                pfb[s] = *reinterpret_cast<const float4*>(wrow + kn + s * 8 + 4);
            }
        }

        const unsigned short* Ac = As[cur];
        const unsigned short* Bc = Bs[cur];
        bf16x16 b = lds_frag(Bc, wv * 16 + rA, k0);
#pragma unroll
        for (int mi = 0; mi < 4; ++mi) {
            bf16x16 a = lds_frag(Ac, mi * 16 + rA, k0);
            acc[mi] = __builtin_amdgcn_wmma_f32_16x16x32_bf16(false, a, false, b,
                                                              (short)0, acc[mi], false, false);
        }

        if (more) {
#if !USE_ASYNC_LDS
            *reinterpret_cast<uint4*>(&As[nxt][arow * LDS_PAD + aseg]) = areg;
#endif
#pragma unroll
            for (int s = 0; s < 2; ++s) {
                *reinterpret_cast<uint4*>(&Bs[nxt][brow * LDS_PAD + bseg + s * 8]) =
                    make_uint4(pack2(pfa[s].x, pfa[s].y), pack2(pfa[s].z, pfa[s].w),
                               pack2(pfb[s].x, pfb[s].y), pack2(pfb[s].z, pfb[s].w));
            }
#if USE_ASYNC_LDS
            WAIT_ASYNC();
#endif
            __syncthreads();
        }
        cur = nxt;
    }

    const int h     = n0 + wv * 16 + (lane & 15);
    const int mbase = m0 + ((lane < 16) ? 0 : 8);
#pragma unroll
    for (int mi = 0; mi < 4; ++mi) {
#pragma unroll
        for (int j = 0; j < 8; ++j) {
            int t   = mbase + mi * 16 + j;
            float c = combine ? combine[(size_t)t * NEXP + expert] : 1.0f;
            size_t o = (size_t)t * HDIM + h;
            out[o] += c * acc[mi][j];
        }
    }
}

// ================= host-side orchestration =================
extern "C" void kernel_launch(void* const* d_in, const int* in_sizes, int n_in,
                              void* d_out, int out_size, void* d_ws, size_t ws_size,
                              hipStream_t stream) {
    const float* x      = (const float*)d_in[0];  // [1,T,H]
    const float* gate_w = (const float*)d_in[1];  // [E,H]
    const float* w13    = (const float*)d_in[2];  // [E,2I,H]
    const float* w2     = (const float*)d_in[3];  // [E,H,I]
    const float* sw13   = (const float*)d_in[4];  // [2*IS,H]
    const float* sw2    = (const float*)d_in[5];  // [H,IS]
    float* out          = (float*)d_out;          // [1,T,H]

    // workspace layout (~19.7 MB): combine | x_bf16 | act_bf16
    float* combine          = (float*)d_ws;
    unsigned short* xb      = (unsigned short*)((char*)d_ws + (size_t)T_TOK * NEXP * 4);
    unsigned short* act     = xb + (size_t)T_TOK * HDIM;

    const int n4 = T_TOK * HDIM / 4;

    moe_gate<<<T_TOK / 8, 256, 0, stream>>>(x, gate_w, combine);
    cvt_bf16<<<(n4 + 255) / 256, 256, 0, stream>>>((const float4*)x, (uint2*)xb, n4);
    zero_f32<<<(n4 + 255) / 256, 256, 0, stream>>>((float4*)out, n4);

    dim3 blk(256);
    dim3 g1(IDIM / 128, T_TOK / 64);   // 11 x 32
    dim3 g2(HDIM / 128, T_TOK / 64);   // 16 x 32
    for (int e = 0; e < NEXP; ++e) {
        const float* w13e = w13 + (size_t)e * 2 * IDIM * HDIM;
        const float* w2e  = w2 + (size_t)e * HDIM * IDIM;
        moe_gemm1<<<g1, blk, 0, stream>>>(xb, w13e, act, HDIM, IDIM, IDIM);
        moe_gemm2<<<g2, blk, 0, stream>>>(act, w2e, combine, e, out, IDIM);
    }
    // shared expert (combine == nullptr -> weight 1.0)
    dim3 gs1(ISDIM / 128, T_TOK / 64); // 22 x 32
    moe_gemm1<<<gs1, blk, 0, stream>>>(xb, sw13, act, HDIM, ISDIM, ISDIM);
    moe_gemm2<<<g2, blk, 0, stream>>>(act, sw2, nullptr, 0, out, ISDIM);
}
